// DNAVaultParadox_57312043598258
// MI455X (gfx1250) — matrix operand order repaired and checked by reading, not compiled, not run on hardware
//
#include <hip/hip_runtime.h>

typedef __attribute__((ext_vector_type(16))) _Float16 v16h;
typedef __attribute__((ext_vector_type(8)))  _Float16 v8h;
typedef __attribute__((ext_vector_type(8)))  float    v8f;
typedef __attribute__((ext_vector_type(4)))  float    v4f;

#define BDIM     256
#define D_       128
#define H_       512
#define E_       2
#define BM       128          // x rows per tile
#define TPB      2            // x tiles per block (w1 staged once, used TPB times)
#define LSTRIDE  136          // halves per padded LDS row (272B stride -> conflict-free b128)
#define NROWS_W1 (E_ * H_)    // 1024 rows (both experts)

// LDS: w1 278528 + x 34816 + outstage 4096 + tsel 512 = 317952 B <= 327680 (320KB)

__global__ __launch_bounds__(BDIM)
void moe2_wmma_kernel(const float* __restrict__ x,
                      const float* __restrict__ w1,
                      const float* __restrict__ b1,
                      const float* __restrict__ w2,
                      const float* __restrict__ b2,
                      const float* __restrict__ prot,
                      float* __restrict__ out)
{
    extern __shared__ char smem_raw[];
    _Float16* w1h      = (_Float16*)smem_raw;                 // [1024][136] f16 (persistent)
    _Float16* xh       = w1h + NROWS_W1 * LSTRIDE;            // [128][136]  f16 (per tile)
    float*    outstage = (float*)(xh + BM * LSTRIDE);         // [E][2 mhalf][128][2] f32
    int*      tsel     = (int*)(outstage + E_ * 2 * BM * 2);  // [128]

    const int tid  = threadIdx.x;
    const int lane = tid & 31;
    const int wave = tid >> 5;

    // ---------------- Stage w1 into LDS as f16 (once per block) ----------------
    for (int r = tid; r < NROWS_W1; r += BDIM) {
        const float4* src = (const float4*)(w1 + (size_t)r * D_);
        _Float16* dst = w1h + r * LSTRIDE;
        #pragma unroll
        for (int q = 0; q < D_ / 4; ++q) {
            float4 v = src[q];
            dst[q * 4 + 0] = (_Float16)v.x;
            dst[q * 4 + 1] = (_Float16)v.y;
            dst[q * 4 + 2] = (_Float16)v.z;
            dst[q * 4 + 3] = (_Float16)v.w;
        }
    }

    // wave role (fixed across tiles)
    const int half = lane >> 4;        // 0: lanes 0-15, 1: lanes 16-31
    const int l16  = lane & 15;
    const int e     = wave >> 2;       // expert
    const int mhalf = (wave >> 1) & 1; // which half of the 32 h-tiles
    const int npair = wave & 1;        // which 4 of the 8 batch-row tiles
    const int nbase0 = npair * 64;
    const int mt0    = mhalf * 16;

    for (int t = 0; t < TPB; ++t) {
        const long rowTile = ((long)blockIdx.x * TPB + t) * BM;

        // ---------------- Stage x tile into LDS as f16 ----------------
        {
            const int r  = tid >> 1;
            const int hs = tid & 1;
            const float4* src = (const float4*)(x + (rowTile + r) * D_ + hs * 64);
            _Float16* dst = xh + r * LSTRIDE + hs * 64;
            #pragma unroll
            for (int q = 0; q < 16; ++q) {
                float4 v = src[q];
                dst[q * 4 + 0] = (_Float16)v.x;
                dst[q * 4 + 1] = (_Float16)v.y;
                dst[q * 4 + 2] = (_Float16)v.z;
                dst[q * 4 + 3] = (_Float16)v.w;
            }
        }
        __syncthreads();

        // ---------------- Routing: argmin over squared prototype distances ----------------
        if (tid < BM) {
            const _Float16* xr = xh + tid * LSTRIDE;
            float d0 = 0.f, d1 = 0.f;
            for (int k = 0; k < D_; ++k) {      // prototype loads are wave-uniform -> scalar
                float xv = (float)xr[k];
                float a0 = xv - prot[k];
                float a1 = xv - prot[D_ + k];
                d0 = fmaf(a0, a0, d0);
                d1 = fmaf(a1, a1, d1);
            }
            tsel[tid] = (d1 < d0) ? 1 : 0;      // first-min ties -> 0 like jnp.argmin
        }
        __syncthreads();

        // ---------------- Phase 1: A = w1 tile (M = h cols), B = x rows (N = batch) ----------------
        // B registers: 4 N-tiles x 4 K-chunks. B layout: lane = batch column n = l16,
        // lane<16 holds K=0..15 contiguous, lane>=16 holds K=16..31 (per 32-K chunk).
        v16h Bx[4][4];
        #pragma unroll
        for (int nt = 0; nt < 4; ++nt) {
            const _Float16* bp = xh + (nbase0 + nt * 16 + l16) * LSTRIDE + half * 16;
            #pragma unroll
            for (int kc = 0; kc < 4; ++kc) {
                v8h lo = *(const v8h*)(bp + kc * 32);
                v8h hi = *(const v8h*)(bp + kc * 32 + 8);
                #pragma unroll
                for (int i = 0; i < 8; ++i) { Bx[nt][kc][i] = lo[i]; Bx[nt][kc][8 + i] = hi[i]; }
            }
        }

        float pacc0[4], pacc1[4];
        #pragma unroll
        for (int nt = 0; nt < 4; ++nt) { pacc0[nt] = 0.f; pacc1[nt] = 0.f; }

        for (int mt = mt0; mt < mt0 + 16; ++mt) {
            // A = 16 w1 rows (h columns). A layout: lane = row m = l16,
            // lane<16 K=0..15, lane>=16 K=16..31 per 32-K chunk.
            const _Float16* ap = w1h + (e * H_ + mt * 16 + l16) * LSTRIDE + half * 16;

            v8f c[4];
            #pragma unroll
            for (int nt = 0; nt < 4; ++nt) c[nt] = (v8f){};

            #pragma unroll
            for (int kc = 0; kc < 4; ++kc) {
                v8h lo = *(const v8h*)(ap + kc * 32);
                v8h hi = *(const v8h*)(ap + kc * 32 + 8);
                v16h A;
                #pragma unroll
                for (int i = 0; i < 8; ++i) { A[i] = lo[i]; A[8 + i] = hi[i]; }
                #pragma unroll
                for (int nt = 0; nt < 4; ++nt)   // one A tile feeds 4 WMMAs
                    c[nt] = __builtin_amdgcn_wmma_f32_16x16x32_f16(
                                false, A, false, Bx[nt][kc], (short)0, c[nt], false, false);
            }

            // per-lane h columns: hcol = mt*16 + half*8 + r  (half-uniform -> broadcast loads)
            const int hc = mt * 16 + half * 8;
            v4f b1a = *(const v4f*)(b1 + e * H_ + hc);
            v4f b1b = *(const v4f*)(b1 + e * H_ + hc + 4);
            v4f w0a = *(const v4f*)(w2 + (e * 2 + 0) * H_ + hc);
            v4f w0b = *(const v4f*)(w2 + (e * 2 + 0) * H_ + hc + 4);
            v4f w1a = *(const v4f*)(w2 + (e * 2 + 1) * H_ + hc);
            v4f w1b = *(const v4f*)(w2 + (e * 2 + 1) * H_ + hc + 4);
            float bias[8], w20[8], w21[8];
            #pragma unroll
            for (int i = 0; i < 4; ++i) {
                bias[i] = b1a[i]; bias[4 + i] = b1b[i];
                w20[i]  = w0a[i]; w20[4 + i]  = w0b[i];
                w21[i]  = w1a[i]; w21[4 + i]  = w1b[i];
            }

            #pragma unroll
            for (int nt = 0; nt < 4; ++nt) {
                #pragma unroll
                for (int r = 0; r < 8; ++r) {
                    float hv = fmaxf(c[nt][r] + bias[r], 0.f);
                    pacc0[nt] = fmaf(hv, w20[r], pacc0[nt]);
                    pacc1[nt] = fmaf(hv, w21[r], pacc1[nt]);
                }
            }
        }

        // combine the two 16-lane halves (complementary h columns, same batch row)
        #pragma unroll
        for (int nt = 0; nt < 4; ++nt) {
            pacc0[nt] += __shfl_xor(pacc0[nt], 16, 32);
            pacc1[nt] += __shfl_xor(pacc1[nt], 16, 32);
        }
        if (half == 0) {
            #pragma unroll
            for (int nt = 0; nt < 4; ++nt) {
                const int row = nbase0 + nt * 16 + l16;
                float* slot = outstage + (((e * 2 + mhalf) * BM) + row) * 2;
                slot[0] = pacc0[nt];      // each slot written by exactly one wave
                slot[1] = pacc1[nt];
            }
        }
        __syncthreads();

        // ---------------- Final: combine m-halves, select expert, add b2, store ----------------
        if (tid < BM) {
            const int eS = tsel[tid];
            const float* s0 = outstage + (((eS * 2 + 0) * BM) + tid) * 2;
            const float* s1 = outstage + (((eS * 2 + 1) * BM) + tid) * 2;
            out[(rowTile + tid) * 2 + 0] = s0[0] + s1[0] + b2[eS * 2 + 0];
            out[(rowTile + tid) * 2 + 1] = s0[1] + s1[1] + b2[eS * 2 + 1];
        }
        __syncthreads();   // xh / outstage / tsel reused by next tile
    }
}

extern "C" void kernel_launch(void* const* d_in, const int* in_sizes, int n_in,
                              void* d_out, int out_size, void* d_ws, size_t ws_size,
                              hipStream_t stream) {
    const float* x    = (const float*)d_in[0];
    const float* w1   = (const float*)d_in[1];
    const float* b1   = (const float*)d_in[2];
    const float* w2   = (const float*)d_in[3];
    const float* b2   = (const float*)d_in[4];
    const float* prot = (const float*)d_in[5];
    float* out = (float*)d_out;

    const int Btot    = in_sizes[0] / D_;     // 131072
    const int nblocks = Btot / (BM * TPB);    // 512

    const size_t lds_bytes =
        (size_t)(NROWS_W1 + BM) * LSTRIDE * sizeof(_Float16)   // w1h + xh
        + (size_t)(E_ * 2 * BM * 2) * sizeof(float)            // outstage
        + (size_t)BM * sizeof(int);                            // tsel

    moe2_wmma_kernel<<<nblocks, BDIM, lds_bytes, stream>>>(x, w1, b1, w2, b2, prot, out);
}
